// GOAT_22926535426406
// MI455X (gfx1250) — matrix operand order repaired and checked by reading, compiled, not run on hardware
//
#include <hip/hip_runtime.h>

// MI455X / gfx1250 fused graph-attention kernel.
// One workgroup (256 threads = 8 wave32) per batch element.
// sim = tanh(S @ T^T) via v_wmma_f32_16x16x32_f16 + hardware v_tanh_f32.

typedef __attribute__((ext_vector_type(4)))  float    v4f;
typedef __attribute__((ext_vector_type(8)))  float    v8f;
typedef __attribute__((ext_vector_type(4)))  _Float16 v4h;
typedef __attribute__((ext_vector_type(8)))  _Float16 v8h;
typedef __attribute__((ext_vector_type(16))) _Float16 v16h;

#define BATCH 8192
#define KDIM  64      // neighbors
#define DDIM  64      // embedding dim
#define LDH   72      // f16 LDS row stride (halfs): conflict-free b128 frag loads
#define LDSIM 68      // f32 sim row stride (floats): rows 16B-aligned for b128

// Hardware tanh (single TRANS instruction on CDNA5).
__device__ __forceinline__ float hw_tanhf(float x) {
#if __has_builtin(__builtin_amdgcn_tanhf)
    return __builtin_amdgcn_tanhf(x);
#else
    float y;
    // v_nop satisfies the TRANS-op result hazard (ISA §7.4: 1 indep op after trans).
    asm volatile("v_tanh_f32 %0, %1\n\tv_nop" : "=v"(y) : "v"(x));
    return y;
#endif
}

__global__ __launch_bounds__(256) void GOAT_graph_attn_wmma(
    const float* __restrict__ W,      // [100001, 64]
    const float* __restrict__ smask,  // [B, 64]
    const float* __restrict__ tmask,  // [B, 64]
    const int*   __restrict__ s_nh,   // [B, 64]
    const int*   __restrict__ t_nh,   // [B, 64]
    float*       __restrict__ out)    // [2, B, 64]
{
    __shared__ _Float16 sh_s[KDIM][LDH];
    __shared__ _Float16 sh_t[KDIM][LDH];
    __shared__ float    sim[KDIM][LDSIM];
    __shared__ float    redn[2][KDIM];
    __shared__ float    att[2][KDIM];

    const int b    = blockIdx.x;
    const int tid  = threadIdx.x;
    const int wave = tid >> 5;
    const int lane = tid & 31;
    const int lh   = lane >> 4;   // half-lane group
    const int lm   = lane & 15;

    // ---------------- Phase 1: gather 128 rows, f32 -> f16 into LDS ----------
    // 128 rows * 16 float4 chunks = 2048 chunks; 8 per thread, coalesced.
    // Branchless padding_idx: always load (row 0 exists), then select-zero.
    #pragma unroll
    for (int i = 0; i < 8; ++i) {
        int e4  = tid + i * 256;          // 0..2047
        int row = e4 >> 4;                // 0..127
        int col = (e4 & 15) << 2;         // 0,4,..,60
        int idx = (row < KDIM) ? s_nh[(size_t)b * KDIM + row]
                               : t_nh[(size_t)b * KDIM + (row - KDIM)];
        v4f v = *(const v4f*)(W + (size_t)idx * DDIM + col);
        float keep = (idx != 0) ? 1.0f : 0.0f;   // padding row -> 0
        v4h h;
        h[0] = (_Float16)(v[0] * keep); h[1] = (_Float16)(v[1] * keep);
        h[2] = (_Float16)(v[2] * keep); h[3] = (_Float16)(v[3] * keep);
        _Float16* dst = (row < KDIM) ? &sh_s[row][col] : &sh_t[row - KDIM][col];
        *(v4h*)dst = h;
    }
    __syncthreads();

    // ---------------- Phase 2: sim tiles via WMMA + hardware tanh ------------
    // 16 tiles of 16x16; each wave owns 2 tiles (uniform control flow, EXEC all-1s).
    #pragma unroll
    for (int t = wave * 2; t < wave * 2 + 2; ++t) {
        const int ti = t >> 2;            // tile row (source rows)
        const int tj = t & 3;             // tile col (target rows)
        const int ar = ti * 16 + lm;      // A: source row for this lane
        const int br = tj * 16 + lm;      // B: target row == B-matrix column
        v8f acc = {};
        #pragma unroll
        for (int kb = 0; kb < DDIM; kb += 32) {
            // A 16x32 f16 layout: lane(m,h): K in [kb+8h, +8) and [kb+16+8h, +8)
            v8h alo = *(const v8h*)&sh_s[ar][kb + 8 * lh];
            v8h ahi = *(const v8h*)&sh_s[ar][kb + 16 + 8 * lh];
            // B 32x16 f16 layout: lane(q,h): K in [kb+8h, +8) and [kb+16+8h, +8)
            v8h blo = *(const v8h*)&sh_t[br][kb + 8 * lh];
            v8h bhi = *(const v8h*)&sh_t[br][kb + 16 + 8 * lh];
            v16h a, bm;
            #pragma unroll
            for (int x = 0; x < 8; ++x) {
                a[x] = alo[x];  a[8 + x] = ahi[x];
                bm[x] = blo[x]; bm[8 + x] = bhi[x];
            }
            acc = __builtin_amdgcn_wmma_f32_16x16x32_f16(
                /*neg_a=*/false, a, /*neg_b=*/false, bm,
                /*c_mod=*/(short)0, acc, /*reuse_a=*/false, /*reuse_b=*/false);
        }
        // C/D layout: lanes 0-15: M=v, N=lane; lanes 16-31: M=8+v, N=lane-16
        #pragma unroll
        for (int v = 0; v < 8; ++v) {
            int r = ti * 16 + v + 8 * lh;
            int c = tj * 16 + lm;
            sim[r][c] = hw_tanhf(acc[v]);
        }
    }
    __syncthreads();

    // ---------------- Phase 3: deterministic means + masks ------------------
    if (tid < KDIM) {
        // Row means: contiguous, 16B-aligned rows -> ds_load_b128 chunks.
        const int k = tid;
        const v4f* rp = (const v4f*)&sim[k][0];
        v4f s4 = {0.f, 0.f, 0.f, 0.f};
        #pragma unroll
        for (int c4 = 0; c4 < KDIM / 4; ++c4) s4 += rp[c4];
        float s = s4[0] + s4[1] + s4[2] + s4[3];
        redn[0][k] = s * (1.0f / 64.0f) + smask[(size_t)b * KDIM + k];
    } else if (tid < 2 * KDIM) {
        // Column means: lane q reads column q (consecutive banks across lanes).
        const int q = tid - KDIM;
        float s = 0.f;
        #pragma unroll 8
        for (int k = 0; k < KDIM; ++k) s += sim[k][q];
        redn[1][q] = s * (1.0f / 64.0f) + tmask[(size_t)b * KDIM + q];
    }
    __syncthreads();

    // ---------------- Phase 4: softmax (wave 0 -> source, wave 1 -> target) --
    if (wave < 2) {
        float v0 = redn[wave][lane];
        float v1 = redn[wave][lane + 32];
        float m = fmaxf(v0, v1);
        #pragma unroll
        for (int off = 16; off > 0; off >>= 1)
            m = fmaxf(m, __shfl_xor(m, off, 32));
        float e0 = __expf(v0 - m), e1 = __expf(v1 - m);
        float s = e0 + e1;
        #pragma unroll
        for (int off = 16; off > 0; off >>= 1)
            s += __shfl_xor(s, off, 32);
        float inv = 1.0f / s;
        att[wave][lane]      = e0 * inv;
        att[wave][lane + 32] = e1 * inv;
    }
    __syncthreads();

    // ---------------- Phase 5: attention-weighted reps ----------------------
    if (tid < 2 * KDIM) {
        const int which = tid >> 6;   // 0 = source, 1 = target
        const int d     = tid & 63;
        const _Float16 (*sh)[LDH] = which ? sh_t : sh_s;
        float acc = 0.f;
        #pragma unroll 8
        for (int k = 0; k < KDIM; ++k)
            acc += att[which][k] * (float)sh[k][d];
        out[(size_t)which * ((size_t)BATCH * DDIM) + (size_t)b * DDIM + d] = acc;
    }
}

extern "C" void kernel_launch(void* const* d_in, const int* in_sizes, int n_in,
                              void* d_out, int out_size, void* d_ws, size_t ws_size,
                              hipStream_t stream) {
    // setup_inputs order: W, source_mask, target_mask, source, target, source_nh, target_nh
    const float* W     = (const float*)d_in[0];
    const float* smask = (const float*)d_in[1];
    const float* tmask = (const float*)d_in[2];
    // d_in[3] (source) and d_in[4] (target) feed unused source_emb/target_emb.
    const int* snh = (const int*)d_in[5];
    const int* tnh = (const int*)d_in[6];
    (void)in_sizes; (void)n_in; (void)out_size; (void)d_ws; (void)ws_size;

    GOAT_graph_attn_wmma<<<BATCH, 256, 0, stream>>>(W, smask, tmask, snh, tnh,
                                                    (float*)d_out);
}
